// GraphConvolution_8203387535720
// MI455X (gfx1250) — matrix operand order repaired and checked by reading, not compiled, not run on hardware
//
#include <hip/hip_runtime.h>
#include <hip/hip_bf16.h>

typedef __attribute__((ext_vector_type(16))) _Float16 v16h;
typedef __attribute__((ext_vector_type(8)))  _Float16 v8h;
typedef __attribute__((ext_vector_type(4)))  _Float16 v4h;
typedef __attribute__((ext_vector_type(8)))  float    v8f;

constexpr int DIN   = 512;   // K
constexpr int DOUT  = 512;   // per-weight output dim
constexpr int NCOMB = 1024;  // [W0|W1] combined N
constexpr int BM = 128, BN = 128, BK = 32;
constexpr int LDA = 40;      // padded LDS stride (f16 elems): 80B rows, 16B aligned
constexpr int LDB = 40;

// ---------------------------------------------------------------------------
// Fused dual GEMM: pre[M][1024] = f16(x[M][512]) @ f16([W0|W1][512][1024]),
// fp32 accumulate via v_wmma_f32_16x16x32_f16.
// Block: 256 threads = 8 waves, tile 128x128, waves arranged 2(M) x 4(N),
// each wave computes 4x2 WMMA tiles of 16x16.
// ---------------------------------------------------------------------------
__global__ __launch_bounds__(256)
void gemm_dual_f16(const float* __restrict__ x,
                   const float* __restrict__ W0,
                   const float* __restrict__ W1,
                   float* __restrict__ pre, int M)
{
    __shared__ _Float16 lds_a[BM * LDA];   // [m][k], k-contiguous
    __shared__ _Float16 lds_b[BN * LDB];   // [n][k], k-contiguous (transposed W)

    const int tid   = threadIdx.x;
    const int lane  = tid & 31;
    const int wave  = tid >> 5;      // 0..7
    const int wm    = wave & 1;      // wave row (64 rows each)
    const int wn    = wave >> 1;     // wave col (32 cols each)
    const int bm    = blockIdx.x;
    const int bn    = blockIdx.y;

    const int l16   = lane & 15;
    const int lhalf = lane >> 4;     // 0 or 1

    // W0/W1 selection is uniform per block: BN==128 divides DOUT==512.
    const float* __restrict__ Wsrc = (bn < 4) ? W0 : W1;
    const int ncol0 = ((bn < 4) ? bn : bn - 4) * BN;

    const bool fullM = (bm * BM + BM <= M);

    v8f acc[4][2];
#pragma unroll
    for (int mt = 0; mt < 4; ++mt)
#pragma unroll
        for (int nt = 0; nt < 2; ++nt)
#pragma unroll
            for (int q = 0; q < 8; ++q) acc[mt][nt][q] = 0.0f;

    for (int k0 = 0; k0 < DIN; k0 += BK) {
        // ---- fill A tile: 128 rows x 32 k ----
        if (fullM) {
            // branch-free fast path: float4 loads, packed v4h ds_store_b64
#pragma unroll
            for (int p = 0; p < 4; ++p) {
                const int i  = tid + p * 256;       // float4 index, 0..1023
                const int m  = i >> 3;              // 8 float4 per row
                const int k4 = (i & 7) * 4;
                const float4 f = *(const float4*)&x[(size_t)(bm * BM + m) * DIN + k0 + k4];
                const v4h h = { (_Float16)f.x, (_Float16)f.y,
                                (_Float16)f.z, (_Float16)f.w };
                *(v4h*)&lds_a[m * LDA + k4] = h;
            }
        } else {
            // tail block: clamped address + post-load select (no divergent branch)
            for (int i = tid; i < BM * BK; i += 256) {
                const int m = i >> 5, k = i & 31;
                const int row = bm * BM + m;
                const int rc  = (row < M) ? row : (M - 1);
                float v = x[(size_t)rc * DIN + (k0 + k)];
                v = (row < M) ? v : 0.0f;
                lds_a[m * LDA + k] = (_Float16)v;
            }
        }
        // ---- fill B tile transposed [n][k]: 4 consecutive k per thread,
        //      each k-plane fully coalesced across lanes (consecutive n) ----
#pragma unroll
        for (int p = 0; p < 4; ++p) {
            const int i  = tid + p * 256;           // 0..1023
            const int n  = i & 127;
            const int kq = i >> 7;                  // 0..7 (k = kq*4)
            const float* wp = &Wsrc[(size_t)(k0 + kq * 4) * DOUT + ncol0 + n];
            const v4h h = { (_Float16)wp[0 * DOUT], (_Float16)wp[1 * DOUT],
                            (_Float16)wp[2 * DOUT], (_Float16)wp[3 * DOUT] };
            *(v4h*)&lds_b[n * LDB + kq * 4] = h;
        }
        __syncthreads();

#pragma unroll
        for (int nt = 0; nt < 2; ++nt) {
            // B frag (32x16 f16): lane holds col n=l16, k = {base..base+7, 16+base..16+base+7}, base = lhalf*8
            const _Float16* bp = &lds_b[(wn * 32 + nt * 16 + l16) * LDB + lhalf * 8];
            const v8h b0 = *(const v8h*)(bp);
            const v8h b1 = *(const v8h*)(bp + 16);
            const v16h bfrag = __builtin_shufflevector(b0, b1,
                0,1,2,3,4,5,6,7,8,9,10,11,12,13,14,15);
#pragma unroll
            for (int mt = 0; mt < 4; ++mt) {
                // A frag (16x32 f16): lane holds row m=l16,
                // k = {base..base+7, 16+base..16+base+7}, base = lhalf*8
                const _Float16* ap = &lds_a[(wm * 64 + mt * 16 + l16) * LDA + lhalf * 8];
                const v8h a0 = *(const v8h*)(ap);
                const v8h a1 = *(const v8h*)(ap + 16);
                const v16h afrag = __builtin_shufflevector(a0, a1,
                    0,1,2,3,4,5,6,7,8,9,10,11,12,13,14,15);
                acc[mt][nt] = __builtin_amdgcn_wmma_f32_16x16x32_f16(
                    false, afrag, false, bfrag, (short)0, acc[mt][nt], false, false);
            }
        }
        __syncthreads();
    }

    // ---- store: C layout — VGPR r: lanes0-15 M=r, lanes16-31 M=8+r; N=l16 ----
#pragma unroll
    for (int mt = 0; mt < 4; ++mt) {
        const int rowb = bm * BM + wm * 64 + mt * 16 + lhalf * 8;
#pragma unroll
        for (int nt = 0; nt < 2; ++nt) {
            const int col = bn * BN + wn * 32 + nt * 16 + l16;
#pragma unroll
            for (int r = 0; r < 8; ++r) {
                const int row = rowb + r;
                if (row < M)
                    pre[(size_t)row * NCOMB + col] = acc[mt][nt][r];
            }
        }
    }
}

// ---------------------------------------------------------------------------
// COO SpMM scatter: one wave32 per edge; f32 atomics at L2.
// out[row, :] += val * pre[col, colOff : colOff+512]
// ---------------------------------------------------------------------------
__global__ __launch_bounds__(256)
void spmm_atomic(const float* __restrict__ pre,
                 const float* __restrict__ vals,
                 const int*   __restrict__ rows,
                 const int*   __restrict__ cols,
                 float* __restrict__ out, int E, int colOff)
{
    const int e = blockIdx.x * 8 + (threadIdx.x >> 5);
    if (e >= E) return;
    const int lane = threadIdx.x & 31;

    const float v = vals[e];
    const int   r = rows[e];
    const int   c = cols[e];

    const float* __restrict__ src = pre + (size_t)c * NCOMB + colOff;
    float* __restrict__ dst = out + (size_t)r * DOUT;

#pragma unroll
    for (int j = lane * 4; j < DOUT; j += 128) {
        const float4 p = *(const float4*)(src + j);
        unsafeAtomicAdd(dst + j + 0, v * p.x);
        unsafeAtomicAdd(dst + j + 1, v * p.y);
        unsafeAtomicAdd(dst + j + 2, v * p.z);
        unsafeAtomicAdd(dst + j + 3, v * p.w);
    }
}

__global__ __launch_bounds__(256)
void zero_f4(float4* __restrict__ out, int n4)
{
    const int i = blockIdx.x * 256 + threadIdx.x;
    if (i < n4) out[i] = make_float4(0.f, 0.f, 0.f, 0.f);
}

__global__ __launch_bounds__(256)
void bias_relu(float4* __restrict__ out, const float* __restrict__ bias, int n4)
{
    const int i = blockIdx.x * 256 + threadIdx.x;
    if (i >= n4) return;
    const int col = (i * 4) & (DOUT - 1);
    float4 v = out[i];
    v.x = fmaxf(v.x + bias[col + 0], 0.f);
    v.y = fmaxf(v.y + bias[col + 1], 0.f);
    v.z = fmaxf(v.z + bias[col + 2], 0.f);
    v.w = fmaxf(v.w + bias[col + 3], 0.f);
    out[i] = v;
}

extern "C" void kernel_launch(void* const* d_in, const int* in_sizes, int n_in,
                              void* d_out, int out_size, void* d_ws, size_t ws_size,
                              hipStream_t stream)
{
    const float* x     = (const float*)d_in[0];
    const float* W0    = (const float*)d_in[1];
    const float* W1    = (const float*)d_in[2];
    const float* bias  = (const float*)d_in[3];
    const float* vals0 = (const float*)d_in[4];
    const float* vals1 = (const float*)d_in[5];
    const int*   rows0 = (const int*)d_in[6];
    const int*   cols0 = (const int*)d_in[7];
    const int*   rows1 = (const int*)d_in[8];
    const int*   cols1 = (const int*)d_in[9];

    float* out = (float*)d_out;
    float* pre = (float*)d_ws;                 // [M][1024] fp32, ~205 MB

    const int M = in_sizes[0] / DIN;           // 50000
    const int E = in_sizes[4];                 // 800000

    // 1) pre = f16(x) @ f16([W0|W1]) with fp32 accumulate
    dim3 ggrid((M + BM - 1) / BM, NCOMB / BN);
    gemm_dual_f16<<<ggrid, 256, 0, stream>>>(x, W0, W1, pre, M);

    // 2) zero the accumulator (d_out) — harness poisons it
    const int n4 = out_size / 4;
    zero_f4<<<(n4 + 255) / 256, 256, 0, stream>>>((float4*)out, n4);

    // 3) scatter both edge sets with f32 atomics
    const int eb = (E + 7) / 8;
    spmm_atomic<<<eb, 256, 0, stream>>>(pre, vals0, rows0, cols0, out, E, 0);
    spmm_atomic<<<eb, 256, 0, stream>>>(pre, vals1, rows1, cols1, out, E, DOUT);

    // 4) out = relu(out + bias), in place
    bias_relu<<<(n4 + 255) / 256, 256, 0, stream>>>((float4*)out, bias, n4);
}
